// Head_39058432589988
// MI455X (gfx1250) — matrix operand order repaired and checked by reading, not compiled, not run on hardware
//
#include <hip/hip_runtime.h>
#include <hip/hip_bf16.h>

// ---------------------------------------------------------------------------
// Types
// ---------------------------------------------------------------------------
typedef __bf16 bf16_t;
typedef __bf16 v4bf  __attribute__((ext_vector_type(4)));
typedef __bf16 v8bf  __attribute__((ext_vector_type(8)));
typedef __bf16 v16bf __attribute__((ext_vector_type(16)));
typedef float  v8f   __attribute__((ext_vector_type(8)));

#define B_SZ   512
#define T_LEN  256
#define C_DIM  512
#define H_DIM  64
#define RANK   32
#define NOUT   192            // 3 * 64 (q,k,v concatenated)
#define SCALE_LORA 2.0f       // ALPHA / RANK = 64/32
#define ATTN_SCALE 0.125f     // HEAD_SIZE^-0.5

__device__ __forceinline__ bf16_t f2bf(float f) {
    // round-to-nearest-even f32 -> bf16
    unsigned u = __builtin_bit_cast(unsigned, f);
    unsigned r = u + 0x7FFFu + ((u >> 16) & 1u);
    unsigned short h = (unsigned short)(r >> 16);
    return __builtin_bit_cast(bf16_t, h);
}

__device__ __forceinline__ v16bf cat8(v8bf lo, v8bf hi) {
    return __builtin_shufflevector(lo, hi, 0,1,2,3,4,5,6,7,8,9,10,11,12,13,14,15);
}

// ---------------------------------------------------------------------------
// Kernel 1: fold LoRA into effective weights, convert to bf16.
//   W_eff[o][c] = W[o][c] + 2.0 * sum_r B[o][r]*A[r][c],  o in [0,192)
// ---------------------------------------------------------------------------
__global__ void prep_weights(const float* __restrict__ Wq, const float* __restrict__ Aq,
                             const float* __restrict__ Bq,
                             const float* __restrict__ Wk, const float* __restrict__ Ak,
                             const float* __restrict__ Bk,
                             const float* __restrict__ Wv, const float* __restrict__ Av,
                             const float* __restrict__ Bv,
                             bf16_t* __restrict__ weff) {
    int o   = blockIdx.x;          // 0..191
    int sel = o >> 6;              // 0=q,1=k,2=v
    int oo  = o & 63;
    const float* W  = (sel == 0) ? Wq : (sel == 1) ? Wk : Wv;
    const float* A  = (sel == 0) ? Aq : (sel == 1) ? Ak : Av;
    const float* Bm = (sel == 0) ? Bq : (sel == 1) ? Bk : Bv;
    for (int c = threadIdx.x; c < C_DIM; c += blockDim.x) {
        float s = 0.f;
        #pragma unroll
        for (int r = 0; r < RANK; ++r) s += Bm[oo * RANK + r] * A[r * C_DIM + c];
        float v = W[oo * C_DIM + c] + SCALE_LORA * s;
        weff[o * C_DIM + c] = f2bf(v);
    }
}

// ---------------------------------------------------------------------------
// Kernel 2: qkv = x @ W_eff^T.   One wave (32 threads) per 16-row strip of x.
// Loop order: k outer, n inner -> 12 live accumulators, A-fragment loaded once
// per k-step (no spills, 12x less LDS A traffic).
// Output: bf16 [B*T][192]  (q:0-63, k:64-127, v:128-191)
// ---------------------------------------------------------------------------
#define XS_STRIDE 520   // 512 + 8 bf16 pad -> rows map to distinct LDS banks

__global__ void qkv_proj(const float* __restrict__ x,
                         const bf16_t* __restrict__ weff,
                         bf16_t* __restrict__ qkv) {
    __shared__ __align__(32) bf16_t xs[16 * XS_STRIDE];

    const int  lane = threadIdx.x;                 // 0..31 (one wave)
    const long row0 = (long)blockIdx.x * 16;       // global row in [0, B*T)

    // stage 16x512 f32 strip of x into LDS as bf16 (float4 loads, b64 stores)
    #pragma unroll 4
    for (int j = 0; j < 64; ++j) {
        int idx = j * 128 + lane * 4;              // 0..8191
        int r = idx >> 9, c = idx & 511;
        const float4 f = *(const float4*)(x + (row0 + r) * C_DIM + c);
        v4bf pk;
        pk[0] = f2bf(f.x); pk[1] = f2bf(f.y); pk[2] = f2bf(f.z); pk[3] = f2bf(f.w);
        *(v4bf*)(xs + r * XS_STRIDE + c) = pk;
    }
    __syncthreads();
    asm volatile("s_wait_dscnt 0" ::: "memory");

    const int half = lane >> 4;
    const int col  = lane & 15;
    const int kb8  = half * 8;    // added to xa, cancelled in the A loads (A K base is kb16)
    const int kb16 = half * 16;   // B-matrix lane-half K base

    const bf16_t* xa = xs   + col * XS_STRIDE + kb8;   // A row = lane&15
    const bf16_t* wb = weff + col * C_DIM     + kb16;  // B col = lane&15

    v8f acc[12];
    const v8f vzero = {};
    #pragma unroll
    for (int nt = 0; nt < 12; ++nt) acc[nt] = vzero;

    #pragma unroll 2
    for (int k0 = 0; k0 < C_DIM; k0 += 32) {
        v8bf alo = *(const v8bf*)(xa + k0 - kb8 + kb16);      // K = kb16 + 0..7
        v8bf ahi = *(const v8bf*)(xa + k0 - kb8 + kb16 + 8);  // K = kb16 + 8..15
        v16bf a  = cat8(alo, ahi);
        #pragma unroll
        for (int nt = 0; nt < 12; ++nt) {
            v16bf b = *(const v16bf*)(wb + nt * 16 * C_DIM + k0);  // K contiguous
            acc[nt] = __builtin_amdgcn_wmma_f32_16x16x32_bf16(
                          false, a, false, b, (short)0, acc[nt], false, false);
        }
    }

    // C/D layout: row = r + 8*half, col = lane&15
    bf16_t* outp = qkv + (row0 + 8 * half) * NOUT + col;
    #pragma unroll
    for (int nt = 0; nt < 12; ++nt) {
        #pragma unroll
        for (int r = 0; r < 8; ++r)
            outp[r * NOUT + nt * 16] = f2bf(acc[nt][r]);
    }
}

// ---------------------------------------------------------------------------
// Kernel 3: causal flash attention per batch. 8 waves/block; wave w owns
// q-tiles {w, 15-w}. V staged transposed in LDS; K/Q fragments straight from
// the bf16 qkv buffer (contiguous 16-element loads, L2-resident).
// ---------------------------------------------------------------------------
#define VT_STRIDE 272   // 256 + 16 bf16 pad: rows stay 32B aligned, banks spread

__global__ void attention(const bf16_t* __restrict__ qkv,
                          float* __restrict__ out) {
    __shared__ __align__(32) bf16_t vsT[H_DIM * VT_STRIDE];   // V^T  [h][s]
    __shared__ __align__(32) bf16_t pbuf[8 * 16 * 32];        // per-wave P tile

    const int  tid  = threadIdx.x;
    const int  b    = blockIdx.x;
    const long base = (long)b * T_LEN * NOUT;

    // stage V transposed into LDS (bf16, lanes read consecutive h -> coalesced)
    for (int idx = tid; idx < H_DIM * T_LEN; idx += 256) {
        int h = idx & 63;
        int s = idx >> 6;
        vsT[h * VT_STRIDE + s] = qkv[base + (long)s * NOUT + 128 + h];
    }
    __syncthreads();

    const int wid  = tid >> 5;
    const int lane = tid & 31;
    const int half = lane >> 4;
    const int col  = lane & 15;
    const int kb8  = half * 8;
    const int kb16 = half * 16;
    bf16_t* pb = pbuf + wid * (16 * 32);

    for (int qi = 0; qi < 2; ++qi) {
        const int qt   = (qi == 0) ? wid : 15 - wid;   // q-tile index 0..15
        const int qend = qt * 16 + 15;

        // Q fragments for this tile: A row = lane&15 -> t = qt*16 + col
        const bf16_t* qp = qkv + base + (long)(qt * 16 + col) * NOUT;
        v16bf qf[2];
        #pragma unroll
        for (int ks = 0; ks < 2; ++ks) {
            v8bf lo = *(const v8bf*)(qp + ks * 32 + kb16);
            v8bf hi = *(const v8bf*)(qp + ks * 32 + kb16 + 8);
            qf[ks] = cat8(lo, hi);
        }

        float m[8], l[8];
        v8f o0 = {}, o1 = {}, o2 = {}, o3 = {};
        #pragma unroll
        for (int r = 0; r < 8; ++r) { m[r] = -1e30f; l[r] = 0.f; }

        const int nJ = qt / 2 + 1;                     // 32-wide s-blocks needed
        for (int j = 0; j < nJ; ++j) {
            const int s0 = j * 32;
            float sv[2][8];
            float tmax[8];
            #pragma unroll
            for (int r = 0; r < 8; ++r) tmax[r] = -1e30f;

            // ---- S = Q K^T for two 16-wide sub-tiles ----
            #pragma unroll
            for (int st = 0; st < 2; ++st) {
                const int sc0  = s0 + st * 16;
                const bool live = (sc0 <= qend);       // wave-uniform
                v8f sacc = {};
                if (live) {
                    const bf16_t* kp = qkv + base + (long)(sc0 + col) * NOUT + 64 + kb16;
                    #pragma unroll
                    for (int ks = 0; ks < 2; ++ks) {
                        v16bf kf = *(const v16bf*)(kp + ks * 32);
                        sacc = __builtin_amdgcn_wmma_f32_16x16x32_bf16(
                                   false, qf[ks], false, kf, (short)0, sacc, false, false);
                    }
                }
                const int scol = sc0 + col;
                #pragma unroll
                for (int r = 0; r < 8; ++r) {
                    int t = qt * 16 + r + 8 * half;    // C-layout row
                    float v = (live && scol <= t) ? sacc[r] * ATTN_SCALE : -1e30f;
                    sv[st][r] = v;
                    tmax[r] = fmaxf(tmax[r], v);
                }
            }

            // ---- online softmax (f32), half-wave reductions ----
            float p0a[8], p1a[8];
            #pragma unroll
            for (int r = 0; r < 8; ++r) {
                float v = tmax[r];
                #pragma unroll
                for (int msk = 1; msk < 16; msk <<= 1)
                    v = fmaxf(v, __shfl_xor(v, msk, 32));
                float mn = fmaxf(m[r], v);
                float al = __expf(m[r] - mn);
                m[r] = mn;
                float p0 = __expf(sv[0][r] - mn);
                float p1 = __expf(sv[1][r] - mn);
                p0a[r] = p0; p1a[r] = p1;
                float rs = p0 + p1;
                #pragma unroll
                for (int msk = 1; msk < 16; msk <<= 1)
                    rs += __shfl_xor(rs, msk, 32);
                l[r] = l[r] * al + rs;
                o0[r] *= al; o1[r] *= al; o2[r] *= al; o3[r] *= al;
            }

            // ---- P (C-layout) -> LDS -> A-layout fragment ----
            #pragma unroll
            for (int r = 0; r < 8; ++r) {
                pb[(r + 8 * half) * 32 + col]      = f2bf(p0a[r]);
                pb[(r + 8 * half) * 32 + 16 + col] = f2bf(p1a[r]);
            }
            asm volatile("s_wait_dscnt 0" ::: "memory");  // wave-local LDS RAW
            v16bf pf;
            {
                v8bf lo = *(const v8bf*)(pb + col * 32 + kb16);
                v8bf hi = *(const v8bf*)(pb + col * 32 + kb16 + 8);
                pf = cat8(lo, hi);
            }

            // ---- O += P @ V  (4 n-tiles over head dim) ----
            {
                const bf16_t* vb = vsT + col * VT_STRIDE + s0 + kb16;
                v16bf vf;
                vf = *(const v16bf*)(vb + 0 * 16 * VT_STRIDE);
                o0 = __builtin_amdgcn_wmma_f32_16x16x32_bf16(false, pf, false, vf, (short)0, o0, false, false);
                vf = *(const v16bf*)(vb + 1 * 16 * VT_STRIDE);
                o1 = __builtin_amdgcn_wmma_f32_16x16x32_bf16(false, pf, false, vf, (short)0, o1, false, false);
                vf = *(const v16bf*)(vb + 2 * 16 * VT_STRIDE);
                o2 = __builtin_amdgcn_wmma_f32_16x16x32_bf16(false, pf, false, vf, (short)0, o2, false, false);
                vf = *(const v16bf*)(vb + 3 * 16 * VT_STRIDE);
                o3 = __builtin_amdgcn_wmma_f32_16x16x32_bf16(false, pf, false, vf, (short)0, o3, false, false);
            }
        }

        // ---- normalize and write out: out[b][t][h], t = qt*16 + r + 8*half ----
        #pragma unroll
        for (int r = 0; r < 8; ++r) {
            long t = qt * 16 + r + 8 * half;
            float inv = 1.0f / l[r];
            float* orow = out + ((long)b * T_LEN + t) * H_DIM + col;
            orow[0]  = o0[r] * inv;
            orow[16] = o1[r] * inv;
            orow[32] = o2[r] * inv;
            orow[48] = o3[r] * inv;
        }
    }
}

// ---------------------------------------------------------------------------
// Launch
// ---------------------------------------------------------------------------
extern "C" void kernel_launch(void* const* d_in, const int* in_sizes, int n_in,
                              void* d_out, int out_size, void* d_ws, size_t ws_size,
                              hipStream_t stream) {
    (void)in_sizes; (void)n_in; (void)out_size; (void)ws_size;
    const float* x  = (const float*)d_in[0];
    const float* Wq = (const float*)d_in[1];
    const float* Aq = (const float*)d_in[2];
    const float* Bq = (const float*)d_in[3];
    const float* Wk = (const float*)d_in[4];
    const float* Ak = (const float*)d_in[5];
    const float* Bk = (const float*)d_in[6];
    const float* Wv = (const float*)d_in[7];
    const float* Av = (const float*)d_in[8];
    const float* Bv = (const float*)d_in[9];

    bf16_t* weff = (bf16_t*)d_ws;                               // 192*512*2 = 196608 B
    bf16_t* qkv  = (bf16_t*)((char*)d_ws + (size_t)NOUT * C_DIM * sizeof(bf16_t));
                                                                 // B*T*192*2 = 50.3 MB
    float* outp = (float*)d_out;

    prep_weights<<<dim3(NOUT), dim3(128), 0, stream>>>(Wq, Aq, Bq, Wk, Ak, Bk, Wv, Av, Bv, weff);
    qkv_proj   <<<dim3((B_SZ * T_LEN) / 16), dim3(32),  0, stream>>>(x, weff, qkv);
    attention  <<<dim3(B_SZ),               dim3(256), 0, stream>>>(qkv, outp);
}
